// PointTransformerReg_60851096649957
// MI455X (gfx1250) — matrix operand order repaired and checked by reading, not compiled, not run on hardware
//
#include <hip/hip_runtime.h>

// ---------------------------------------------------------------------------
// PointTransformer regressor for MI455X (gfx1250).
// All dense GEMMs run through V_WMMA_F32_16X16X32_F16 (fp16 in, fp32 acc).
// GEMM: 128x64 block tile, double-buffered LDS, b128-vectorized global
// staging, fused epilogues (alpha/BN-scale/bias/ReLU/GELU/residual), batched
// strides for per-(b,h) attention GEMMs.
// ---------------------------------------------------------------------------

typedef _Float16 half_t;
typedef __attribute__((ext_vector_type(16))) _Float16 v16h;
typedef __attribute__((ext_vector_type(8)))  _Float16 h8;
typedef __attribute__((ext_vector_type(2)))  _Float16 h2;
typedef __attribute__((ext_vector_type(8)))  float    v8f;

static constexpr int   cB   = 4;
static constexpr int   cN   = 1024;
static constexpr int   cK   = 16;
static constexpr int   cNP  = 17;
static constexpr int   cDIM = 512;
static constexpr int   cH   = 4;
static constexpr int   cDEPTH = 5;
static constexpr int   cS   = cNP + cN;   // 1041 real tokens
static constexpr int   cSP  = 1056;       // padded token count (mult of 32)
static constexpr int   cBN  = cB * cN;    // 4096 points
static constexpr int   cBNK = cBN * cK;   // 65536 (point,neighbor) rows
static constexpr float cEPS = 1e-5f;
static constexpr float cISQ = 0.08838834764831845f; // 1/sqrt(128) (= SCALE = 1/SQRT_DM)

// ============================= GEMM (WMMA) =================================
// C[m,n] = epilogue( alpha * sum_k A[m,k]*B[k,n] )  ; B optionally transposed.
// Block: 128 threads (4 wave32s). Tile 128(M) x 64(N) x 32(K).
// Each wave owns 32 rows x 64 cols = 8 wmma accumulators; B fragments are
// reused across the two 16-row sub-tiles. LDS tiles are double-buffered:
// global->VGPR loads for tile k+1 issue before the WMMAs of tile k, and are
// committed to the alternate LDS buffer afterwards (one barrier per K-step).
template<int ACT, int TRANSB, int OUTF32>
__global__ __launch_bounds__(128)
void gemm_wmma(const half_t* __restrict__ A, long long lda, long long sAb, long long sAh,
               const half_t* __restrict__ Bm, long long ldb, long long sBb, long long sBh,
               void* __restrict__ Cv, long long ldc, long long sCb, long long sCh,
               const half_t* __restrict__ Res, long long ldr,
               const float* __restrict__ bias, const float* __restrict__ scale,
               int M, int Nn, int Kk, int Hb, float alpha)
{
  __shared__ half_t As[2][128 * 32];   // [m][k] tiles (double buffered)
  __shared__ half_t Bs[2][64 * 32];    // [n][k] tiles (B staged transposed)
  const int tid  = threadIdx.x;
  const int lane = tid & 31;
  const int wv   = tid >> 5;
  const int zb = blockIdx.z / Hb, zh = blockIdx.z % Hb;
  const long long offA = (long long)zb * sAb + (long long)zh * sAh;
  const long long offB = (long long)zb * sBb + (long long)zh * sBh;
  const long long offC = (long long)zb * sCb + (long long)zh * sCh;
  const int tM = blockIdx.y * 128, tN = blockIdx.x * 64;

  v8f acc[2][4] = {};
  h8 aReg[4], bReg[2];

  // global -> VGPR (b128 loads; row-guarded, zero-filled)
  auto loadTile = [&](int kb) {
    #pragma unroll
    for (int i = 0; i < 4; ++i) {
      int chunk = i * 128 + tid;              // 512 chunks cover A 128x32
      int r = chunk >> 2, c8 = (chunk & 3) * 8;
      int m = tM + r;
      h8 v = {};
      if (m < M) v = *(const h8*)&A[offA + (long long)m * lda + (kb + c8)];
      aReg[i] = v;
    }
    #pragma unroll
    for (int i = 0; i < 2; ++i) {
      int chunk = i * 128 + tid;              // 256 chunks cover B 64(n)x32(k)
      h8 v = {};
      if (TRANSB) {
        int n = chunk >> 2, c8 = (chunk & 3) * 8;
        int ng = tN + n;
        if (ng < Nn) v = *(const h8*)&Bm[offB + (long long)ng * ldb + (kb + c8)];
      } else {
        int k = chunk >> 3, n8 = (chunk & 7) * 8;
        int ng = tN + n8;
        if (ng + 8 <= Nn) {
          v = *(const h8*)&Bm[offB + (long long)(kb + k) * ldb + ng];
        } else {
          #pragma unroll
          for (int j = 0; j < 8; ++j)
            v[j] = (ng + j < Nn) ? Bm[offB + (long long)(kb + k) * ldb + (ng + j)]
                                 : (half_t)0.f;
        }
      }
      bReg[i] = v;
    }
  };
  // VGPR -> LDS (A/trans-B: ds_store_b128; non-trans B transposed on store)
  auto storeTile = [&](int bufi) {
    #pragma unroll
    for (int i = 0; i < 4; ++i) {
      int chunk = i * 128 + tid;
      int r = chunk >> 2, c8 = (chunk & 3) * 8;
      *(h8*)&As[bufi][r * 32 + c8] = aReg[i];
    }
    #pragma unroll
    for (int i = 0; i < 2; ++i) {
      int chunk = i * 128 + tid;
      if (TRANSB) {
        int n = chunk >> 2, c8 = (chunk & 3) * 8;
        *(h8*)&Bs[bufi][n * 32 + c8] = bReg[i];
      } else {
        int k = chunk >> 3, n8 = (chunk & 7) * 8;
        #pragma unroll
        for (int j = 0; j < 8; ++j) Bs[bufi][(n8 + j) * 32 + k] = bReg[i][j];
      }
    }
  };

  loadTile(0);
  storeTile(0);
  __syncthreads();

  int buf = 0;
  for (int kb = 0; kb < Kk; kb += 32) {
    const bool hasNext = (kb + 32) < Kk;
    if (hasNext) {
      loadTile(kb + 32);                       // prefetch into VGPRs
      if (kb + 64 < Kk) {                      // L2 prefetch 2 tiles ahead
        int m = tM + (tid >> 1);
        if (m < M) __builtin_prefetch(&A[offA + (long long)m * lda + kb + 64], 0, 1);
      }
    }

    // ---- A fragments (16-bit A 16x32 layout, ISA 7.12.2) ----
    v16h af[2];
    #pragma unroll
    for (int s = 0; s < 2; ++s) {
      int r = (wv << 5) + (s << 4) + (lane & 15);
      #pragma unroll
      for (int i = 0; i < 8; ++i) {
        int kbase = (i < 4) ? ((lane < 16) ? 0 : 8) : ((lane < 16) ? 16 : 24);
        int k2 = kbase + 2 * (i & 3);
        h2 p = *(const h2*)&As[buf][r * 32 + k2];
        af[s][2 * i] = p.x; af[s][2 * i + 1] = p.y;
      }
    }
    const int kbb = (lane < 16) ? 0 : 8;
    #pragma unroll
    for (int ct = 0; ct < 4; ++ct) {
      v16h bf;
      int nrow = ct * 16 + (lane & 15);
      #pragma unroll
      for (int j = 0; j < 8; ++j) {
        h2 p = *(const h2*)&Bs[buf][nrow * 32 + kbb + ((j < 4) ? 0 : 16) + 2 * (j & 3)];
        bf[2 * j] = p.x; bf[2 * j + 1] = p.y;
      }
      #pragma unroll
      for (int s = 0; s < 2; ++s)                 // B fragment reused twice
        acc[s][ct] = __builtin_amdgcn_wmma_f32_16x16x32_f16(
            false, af[s], false, bf, (short)0, acc[s][ct], false, false);
    }

    if (hasNext) { storeTile(buf ^ 1); buf ^= 1; }
    __syncthreads();
  }

  // ---- epilogue (C/D 16x16 f32 layout: VGPR v -> M=v / v+8) ----
  #pragma unroll
  for (int s = 0; s < 2; ++s)
  #pragma unroll
  for (int ct = 0; ct < 4; ++ct)
  #pragma unroll
  for (int v = 0; v < 8; ++v) {
    int m = tM + (wv << 5) + (s << 4) + v + ((lane & 16) ? 8 : 0);
    int n = tN + ct * 16 + (lane & 15);
    if (m < M && n < Nn) {
      float val = alpha * acc[s][ct][v];
      if (scale) val *= scale[n];
      if (bias)  val += bias[n];
      if (ACT == 1) {
        val = fmaxf(val, 0.f);
      } else if (ACT == 2) {  // tanh-approx GELU (jax.nn.gelu default)
        float u = 0.7978845608f * (val + 0.044715f * val * val * val);
        val = 0.5f * val * (1.f + tanhf(u));
      }
      if (Res) val += (float)Res[(long long)m * ldr + n];
      if (OUTF32) ((float*) Cv)[offC + (long long)m * ldc + n] = val;
      else        ((half_t*)Cv)[offC + (long long)m * ldc + n] = (half_t)val;
    }
  }
}

// ============================ helper kernels ===============================

__global__ __launch_bounds__(256) void k_dists(const float* __restrict__ x,
                                               float* __restrict__ d) {
  int e = blockIdx.x * 256 + threadIdx.x;
  if (e >= cB * cN * cN) return;
  int j = e % cN; int r = e / cN; int i = r % cN; int b = r / cN;
  const float* pi = x + (size_t)(b * cN + i) * 5;
  const float* pj = x + (size_t)(b * cN + j) * 5;
  float dx = pi[0] - pj[0], dy = pi[1] - pj[1], dz = pi[2] - pj[2];
  d[e] = dx * dx + dy * dy + dz * dz;
}

// stable 16-NN selection (single pass insertion; strict < keeps earlier index
// on ties == stable argsort semantics).
__global__ __launch_bounds__(128) void k_knn(const float* __restrict__ d,
                                             int* __restrict__ idx) {
  int r = blockIdx.x * 128 + threadIdx.x;
  if (r >= cBN) return;
  const float* row = d + (size_t)r * cN;
  float bd[cK]; int bi[cK];
  #pragma unroll
  for (int k = 0; k < cK; ++k) { bd[k] = 3.4e38f; bi[k] = 0; }
  for (int j = 0; j < cN; ++j) {
    float v = row[j];
    if (v < bd[cK - 1]) {
      int p = cK - 1;
      while (p > 0 && v < bd[p - 1]) { bd[p] = bd[p - 1]; bi[p] = bi[p - 1]; --p; }
      bd[p] = v; bi[p] = j;
    }
  }
  #pragma unroll
  for (int k = 0; k < cK; ++k) idx[(size_t)r * cK + k] = bi[k];
}

__global__ __launch_bounds__(256) void k_cvt_w(const float* __restrict__ src,
                                               half_t* __restrict__ dst,
                                               int Ks, int Ns, int Kp, int Np) {
  int e = blockIdx.x * 256 + threadIdx.x;
  if (e >= Kp * Np) return;
  int n = e % Np, k = e / Np;
  dst[e] = (k < Ks && n < Ns) ? (half_t)src[(size_t)k * Ns + n] : (half_t)0.f;
}

__global__ __launch_bounds__(256) void k_bnprep(const float* g, const float* b,
                                                const float* m, const float* v,
                                                const float* cb,
                                                float* s, float* t, int n) {
  int i = blockIdx.x * 256 + threadIdx.x;
  if (i >= n) return;
  float sc = g[i] * rsqrtf(v[i] + cEPS);
  s[i] = sc;
  t[i] = (cb[i] - m[i]) * sc + b[i];
}

__global__ __launch_bounds__(256) void k_xpad(const float* __restrict__ x,
                                              half_t* __restrict__ xp) {
  int e = blockIdx.x * 256 + threadIdx.x;
  if (e >= cBN * 32) return;
  int c = e & 31, r = e >> 5;
  xp[e] = (c < 5) ? (half_t)x[(size_t)r * 5 + c] : (half_t)0.f;
}

__global__ __launch_bounds__(256) void k_concat(const float* __restrict__ x,
                                                const half_t* __restrict__ f,
                                                half_t* __restrict__ out,
                                                int dp, int Kp) {
  int e = blockIdx.x * 256 + threadIdx.x;
  if (e >= cBN * Kp) return;
  int c = e % Kp, r = e / Kp;
  float v = 0.f;
  if (c < 3) v = x[(size_t)r * 5 + c];
  else if (c < 3 + dp) v = (float)f[(size_t)r * dp + (c - 3)];
  out[e] = (half_t)v;
}

__global__ __launch_bounds__(256) void k_relpad(const float* __restrict__ x,
                                                const int* __restrict__ idx,
                                                half_t* __restrict__ rel) {
  int e = blockIdx.x * 256 + threadIdx.x;
  if (e >= cBNK * 32) return;
  int c = e & 31; int r = e >> 5;            // r = (b*N+m)*K + n
  float v = 0.f;
  if (c < 3) {
    int bm = r / cK; int b = bm / cN;
    int j = idx[r];
    v = x[(size_t)bm * 5 + c] - x[((size_t)b * cN + j) * 5 + c];
  }
  rel[e] = (half_t)v;
}

__global__ __launch_bounds__(256) void k_gather_kv(const half_t* __restrict__ src,
                                                   const int* __restrict__ idx,
                                                   half_t* __restrict__ dst) {
  int e = blockIdx.x * 256 + threadIdx.x;
  if (e >= cBNK * 128) return;
  int f = e & 127; int r = e >> 7;           // r = (b*N+m)*K + n
  int bm = r / cK; int b = bm / cN;
  int j = idx[r];
  dst[e] = src[((size_t)b * cN + j) * 128 + f];
}

__global__ __launch_bounds__(256) void k_a0(const half_t* __restrict__ q,
                                            const half_t* __restrict__ kk,
                                            const half_t* __restrict__ pos,
                                            half_t* __restrict__ a0) {
  int e = blockIdx.x * 256 + threadIdx.x;
  if (e >= cBNK * 128) return;
  int f = e & 127; int r = e >> 7; int bm = r / cK;
  a0[e] = (half_t)((float)q[(size_t)bm * 128 + f] - (float)kk[e] + (float)pos[e]);
}

// softmax over the 16 neighbors (axis=-2) fused with einsum a*(vv+pos) reduce.
__global__ __launch_bounds__(256) void k_tb_reduce(const float* __restrict__ a2,
                                                   const half_t* __restrict__ vv,
                                                   const half_t* __restrict__ pos,
                                                   half_t* __restrict__ res) {
  int e = blockIdx.x * 256 + threadIdx.x;
  if (e >= cBN * 128) return;
  int f = e & 127; int bm = e >> 7;
  size_t base = (size_t)bm * (cK * 128) + f;
  float mx = -3e38f;
  #pragma unroll
  for (int n = 0; n < cK; ++n) mx = fmaxf(mx, a2[base + n * 128] * cISQ);
  float sum = 0.f, accv = 0.f;
  #pragma unroll
  for (int n = 0; n < cK; ++n) {
    size_t o = base + n * 128;
    float p = __expf(a2[o] * cISQ - mx);
    sum += p;
    accv += p * ((float)vv[o] + (float)pos[o]);
  }
  res[(size_t)bm * 128 + f] = (half_t)(accv / sum);
}

__global__ __launch_bounds__(128) void k_layernorm(const half_t* __restrict__ in,
                                                   half_t* __restrict__ out,
                                                   const float* __restrict__ g,
                                                   const float* __restrict__ b,
                                                   int D) {
  int row = blockIdx.x;
  const half_t* x = in + (size_t)row * D;
  half_t* o = out + (size_t)row * D;
  __shared__ float r1[128], r2[128];
  int t = threadIdx.x;
  float s = 0.f, s2 = 0.f;
  for (int j = t; j < D; j += 128) { float v = (float)x[j]; s += v; s2 += v * v; }
  r1[t] = s; r2[t] = s2; __syncthreads();
  for (int w = 64; w > 0; w >>= 1) {
    if (t < w) { r1[t] += r1[t + w]; r2[t] += r2[t + w]; }
    __syncthreads();
  }
  float mu = r1[0] / D;
  float var = fmaxf(r2[0] / D - mu * mu, 0.f);
  float inv = rsqrtf(var + cEPS);
  for (int j = t; j < D; j += 128)
    o[j] = (half_t)(((float)x[j] - mu) * inv * g[j] + b[j]);
}

// masked row softmax (fp32 scores -> fp16 probs, zeros for j>=Sreal).
__global__ __launch_bounds__(128) void k_attn_softmax(const float* __restrict__ sc,
                                                      half_t* __restrict__ pr) {
  size_t row = blockIdx.x;
  const float* s = sc + row * cSP;
  half_t* p = pr + row * cSP;
  __shared__ float red[128];
  int t = threadIdx.x;
  float mx = -3e38f;
  for (int j = t; j < cS; j += 128) mx = fmaxf(mx, s[j]);
  red[t] = mx; __syncthreads();
  for (int w = 64; w > 0; w >>= 1) { if (t < w) red[t] = fmaxf(red[t], red[t + w]); __syncthreads(); }
  mx = red[0]; __syncthreads();
  float sum = 0.f;
  for (int j = t; j < cS; j += 128) sum += __expf(s[j] - mx);
  red[t] = sum; __syncthreads();
  for (int w = 64; w > 0; w >>= 1) { if (t < w) red[t] += red[t + w]; __syncthreads(); }
  float inv = 1.f / red[0];
  for (int j = t; j < cSP; j += 128)
    p[j] = (half_t)((j < cS) ? __expf(s[j] - mx) * inv : 0.f);
}

__global__ __launch_bounds__(256) void k_hinit(const float* __restrict__ jn,
                                               const float* __restrict__ pe,
                                               const half_t* __restrict__ pts,
                                               half_t* __restrict__ h) {
  int e = blockIdx.x * 256 + threadIdx.x;
  if (e >= cB * cSP * cDIM) return;
  int c = e % cDIM; int r = e / cDIM; int s = r % cSP; int b = r / cSP;
  float v = 0.f;
  if (s < cNP)      v = jn[((size_t)b * cNP + s) * cDIM + c] + pe[(size_t)s * cDIM + c];
  else if (s < cS)  v = (float)pts[((size_t)b * cN + (s - cNP)) * cDIM + c];
  h[e] = (half_t)v;
}

__global__ __launch_bounds__(256) void k_gather68(const half_t* __restrict__ h,
                                                  half_t* __restrict__ out) {
  int e = blockIdx.x * 256 + threadIdx.x;
  if (e >= 128 * cDIM) return;
  int c = e % cDIM; int r = e / cDIM;
  half_t v = (half_t)0.f;
  if (r < cB * cNP) {
    int b = r / cNP, j = r % cNP;
    v = h[((size_t)b * cSP + j) * cDIM + c];
  }
  out[e] = v;
}

__global__ __launch_bounds__(256) void k_ycopy(const float* __restrict__ yp,
                                               float* __restrict__ out) {
  int i = blockIdx.x * 256 + threadIdx.x;
  if (i >= cB * cNP * 3) return;
  int b = i / (cNP * 3); int rem = i % (cNP * 3); int r = rem / 3; int c = rem % 3;
  out[i] = yp[(size_t)(b * cNP + r) * 16 + c];
}

// ============================== host side ==================================

static inline int cdiv(int a, int b) { return (a + b - 1) / b; }

static void launch_gemm(hipStream_t st,
                        const half_t* A, long long lda, long long sAb, long long sAh,
                        const half_t* Bm, long long ldb, long long sBb, long long sBh, int transB,
                        void* C, long long ldc, long long sCb, long long sCh, int outF32,
                        const half_t* Res, long long ldr,
                        const float* bias, const float* scale,
                        int M, int Nn, int Kk, int nb, int Hb, float alpha, int act) {
  dim3 g(cdiv(Nn, 64), cdiv(M, 128), nb), b(128);
#define GG(a, t, o) gemm_wmma<a, t, o><<<g, b, 0, st>>>(A, lda, sAb, sAh, Bm, ldb, sBb, sBh, \
    C, ldc, sCb, sCh, Res, ldr, bias, scale, M, Nn, Kk, Hb, alpha)
  int code = act * 4 + transB * 2 + outF32;
  switch (code) {
    case 0:  GG(0, 0, 0); break;  case 1:  GG(0, 0, 1); break;
    case 2:  GG(0, 1, 0); break;  case 3:  GG(0, 1, 1); break;
    case 4:  GG(1, 0, 0); break;  case 5:  GG(1, 0, 1); break;
    case 6:  GG(1, 1, 0); break;  case 7:  GG(1, 1, 1); break;
    case 8:  GG(2, 0, 0); break;  case 9:  GG(2, 0, 1); break;
    case 10: GG(2, 1, 0); break;  default: GG(2, 1, 1); break;
  }
#undef GG
}

static void gemm_nn(hipStream_t st, const half_t* A, int lda, const half_t* W, int ldw,
                    half_t* C, int ldc, const float* bias, const float* scale,
                    int M, int Nn, int Kk, int act,
                    const half_t* Res = nullptr, int ldr = 0) {
  launch_gemm(st, A, lda, 0, 0, W, ldw, 0, 0, 0, C, ldc, 0, 0, 0,
              Res, ldr, bias, scale, M, Nn, Kk, 1, 1, 1.f, act);
}

struct TBW { const float *fc1_w,*fc1_b,*fc2_w,*fc2_b,*d1_w,*d1_b,*d2_w,*d2_b,
                         *g1_w,*g1_b,*g2_w,*g2_b,*wq,*wk,*wv; };
struct TDW { const float *c1_w,*c1_b,*bn1_g,*bn1_b,*bn1_m,*bn1_v,
                         *c2_w,*c2_b,*bn2_g,*bn2_b,*bn2_m,*bn2_v; };
struct LYW { const float *ln1_g,*ln1_b,*qkv_w,*out_w,*out_b,
                         *ln2_g,*ln2_b,*ff1_w,*ff1_b,*ff2_w,*ff2_b; };

extern "C" void kernel_launch(void* const* d_in, const int* in_sizes, int n_in,
                              void* d_out, int out_size, void* d_ws, size_t ws_size,
                              hipStream_t stream) {
  (void)in_sizes; (void)n_in; (void)out_size; (void)ws_size;
  int pi = 0;
  auto P = [&]() { return (const float*)d_in[pi++]; };

  const float* X = P();                       // x (B,N,5)
  const float* fc1a_w = P(); const float* fc1a_b = P();
  const float* fc1b_w = P(); const float* fc1b_b = P();
  auto readTB = [&]() { TBW t;
    t.fc1_w=P(); t.fc1_b=P(); t.fc2_w=P(); t.fc2_b=P();
    t.d1_w=P();  t.d1_b=P();  t.d2_w=P();  t.d2_b=P();
    t.g1_w=P();  t.g1_b=P();  t.g2_w=P();  t.g2_b=P();
    t.wq=P(); t.wk=P(); t.wv=P(); return t; };
  auto readTD = [&]() { TDW t;
    t.c1_w=P(); t.c1_b=P(); t.bn1_g=P(); t.bn1_b=P(); t.bn1_m=P(); t.bn1_v=P();
    t.c2_w=P(); t.c2_b=P(); t.bn2_g=P(); t.bn2_b=P(); t.bn2_m=P(); t.bn2_v=P(); return t; };
  TBW tb0 = readTB();
  TDW td[4]; for (int i = 0; i < 4; ++i) td[i] = readTD();
  TBW tbs[4]; for (int i = 0; i < 4; ++i) tbs[i] = readTB();
  const float* pos_embed = P();
  LYW ly[cDEPTH];
  for (int i = 0; i < cDEPTH; ++i) { LYW l;
    l.ln1_g=P(); l.ln1_b=P(); l.qkv_w=P(); l.out_w=P(); l.out_b=P();
    l.ln2_g=P(); l.ln2_b=P(); l.ff1_w=P(); l.ff1_b=P(); l.ff2_w=P(); l.ff2_b=P();
    ly[i] = l; }
  const float* fc2a_w = P(); const float* fc2a_b = P();
  const float* fc2b_w = P(); const float* fc2b_b = P();
  const float* fc3a_w = P(); const float* fc3a_b = P();
  const float* fc3b_w = P(); const float* fc3b_b = P();
  const float* jn = (const float*)d_in[pi++];  // joint_noise

  // ---------------- workspace bump allocator ----------------
  char* base = (char*)d_ws;
  size_t off = 0;
  auto alloc = [&](size_t bytes) -> void* {
    off = (off + 255) & ~(size_t)255;
    void* p = base + off; off += bytes; return p;
  };

  // persistent region
  half_t* wbuf  = (half_t*)alloc((size_t)512 * 1536 * 2);   // weight cvt scratch
  float*  bnS   = (float*) alloc(512 * 4);
  float*  bnT   = (float*) alloc(512 * 4);
  half_t* fA    = (half_t*)alloc((size_t)cBN * 512 * 2);    // feats ping
  half_t* fB    = (half_t*)alloc((size_t)cBN * 512 * 2);    // feats pong
  half_t* t32   = (half_t*)alloc((size_t)cBN * 32 * 2);
  half_t* xpadb = (half_t*)alloc((size_t)cBN * 32 * 2);
  size_t mark = off;

  // point-cloud phase scratch
  float*  dists = (float*) alloc((size_t)cB * cN * cN * 4);
  int*    idx   = (int*)   alloc((size_t)cBNK * 4);
  half_t* X128  = (half_t*)alloc((size_t)cBN * 128 * 2);
  half_t* Qb    = (half_t*)alloc((size_t)cBN * 128 * 2);
  half_t* Kfb   = (half_t*)alloc((size_t)cBN * 128 * 2);
  half_t* Vfb   = (half_t*)alloc((size_t)cBN * 128 * 2);
  half_t* relp  = (half_t*)alloc((size_t)cBNK * 32 * 2);
  half_t* kkb   = (half_t*)alloc((size_t)cBNK * 128 * 2);
  half_t* vvb   = (half_t*)alloc((size_t)cBNK * 128 * 2);
  half_t* posb  = (half_t*)alloc((size_t)cBNK * 128 * 2);
  half_t* Tb    = (half_t*)alloc((size_t)cBNK * 128 * 2);
  half_t* a0b   = (half_t*)alloc((size_t)cBNK * 128 * 2);
  float*  a2f   = (float*) alloc((size_t)cBNK * 128 * 4);
  half_t* res128= (half_t*)alloc((size_t)cBN * 128 * 2);
  half_t* catb  = (half_t*)alloc((size_t)cBN * 288 * 2);
  half_t* tdt   = (half_t*)alloc((size_t)cBN * 512 * 2);

  auto cvt = [&](const float* src, int Ks, int Ns, int Kp, int Np) {
    int tot = Kp * Np;
    k_cvt_w<<<cdiv(tot, 256), 256, 0, stream>>>(src, wbuf, Ks, Ns, Kp, Np);
  };

  // ---------------- KNN (constant across all tblocks) ----------------
  k_dists<<<cdiv(cB * cN * cN, 256), 256, 0, stream>>>(X, dists);
  k_knn<<<cdiv(cBN, 128), 128, 0, stream>>>(dists, idx);

  // ---------------- initial point MLP: 5 -> 32 -> 32 ----------------
  k_xpad<<<cdiv(cBN * 32, 256), 256, 0, stream>>>(X, xpadb);
  cvt(fc1a_w, 5, 32, 32, 32);
  gemm_nn(stream, xpadb, 32, wbuf, 32, t32, 32, fc1a_b, nullptr, cBN, 32, 32, 1);
  cvt(fc1b_w, 32, 32, 32, 32);
  gemm_nn(stream, t32, 32, wbuf, 32, fA, 32, fc1b_b, nullptr, cBN, 32, 32, 0);

  half_t* cur = fA; half_t* nxt = fB;
  int dp = 32;

  auto run_tblock = [&](const TBW& t) {
    cvt(t.fc1_w, dp, 128, dp, 128);
    gemm_nn(stream, cur, dp, wbuf, 128, X128, 128, t.fc1_b, nullptr, cBN, 128, dp, 0);
    cvt(t.wq, 128, 128, 128, 128);
    gemm_nn(stream, X128, 128, wbuf, 128, Qb, 128, nullptr, nullptr, cBN, 128, 128, 0);
    cvt(t.wk, 128, 128, 128, 128);
    gemm_nn(stream, X128, 128, wbuf, 128, Kfb, 128, nullptr, nullptr, cBN, 128, 128, 0);
    cvt(t.wv, 128, 128, 128, 128);
    gemm_nn(stream, X128, 128, wbuf, 128, Vfb, 128, nullptr, nullptr, cBN, 128, 128, 0);
    k_gather_kv<<<cdiv(cBNK * 128, 256), 256, 0, stream>>>(Kfb, idx, kkb);
    k_gather_kv<<<cdiv(cBNK * 128, 256), 256, 0, stream>>>(Vfb, idx, vvb);
    k_relpad<<<cdiv(cBNK * 32, 256), 256, 0, stream>>>(X, idx, relp);
    cvt(t.d1_w, 3, 128, 32, 128);
    gemm_nn(stream, relp, 32, wbuf, 128, Tb, 128, t.d1_b, nullptr, cBNK, 128, 32, 1);
    cvt(t.d2_w, 128, 128, 128, 128);
    gemm_nn(stream, Tb, 128, wbuf, 128, posb, 128, t.d2_b, nullptr, cBNK, 128, 128, 0);
    k_a0<<<cdiv(cBNK * 128, 256), 256, 0, stream>>>(Qb, kkb, posb, a0b);
    cvt(t.g1_w, 128, 128, 128, 128);
    gemm_nn(stream, a0b, 128, wbuf, 128, Tb, 128, t.g1_b, nullptr, cBNK, 128, 128, 1);
    cvt(t.g2_w, 128, 128, 128, 128);
    launch_gemm(stream, Tb, 128, 0, 0, wbuf, 128, 0, 0, 0,
                a2f, 128, 0, 0, 1, nullptr, 0, t.g2_b, nullptr,
                cBNK, 128, 128, 1, 1, 1.f, 0);
    k_tb_reduce<<<cdiv(cBN * 128, 256), 256, 0, stream>>>(a2f, vvb, posb, res128);
    cvt(t.fc2_w, 128, dp, 128, dp);
    gemm_nn(stream, res128, 128, wbuf, dp, nxt, dp, t.fc2_b, nullptr,
            cBN, dp, 128, 0, cur, dp);
    half_t* tmp = cur; cur = nxt; nxt = tmp;
  };

  run_tblock(tb0);

  static const int couts[4] = {64, 128, 256, 512};
  for (int i = 0; i < 4; ++i) {
    int cin = 3 + dp;
    int Kp = ((cin + 31) / 32) * 32;          // 64, 96, 160, 288
    int cint = couts[i];
    k_concat<<<cdiv(cBN * Kp, 256), 256, 0, stream>>>(X, cur, catb, dp, Kp);
    cvt(td[i].c1_w, cin, cint, Kp, cint);
    k_bnprep<<<cdiv(cint, 256), 256, 0, stream>>>(td[i].bn1_g, td[i].bn1_b, td[i].bn1_m,
                                                  td[i].bn1_v, td[i].c1_b, bnS, bnT, cint);
    gemm_nn(stream, catb, Kp, wbuf, cint, tdt, cint, bnT, bnS, cBN, cint, Kp, 1);
    cvt(td[i].c2_w, cint, cint, cint, cint);
    k_bnprep<<<cdiv(cint, 256), 256, 0, stream>>>(td[i].bn2_g, td[i].bn2_b, td[i].bn2_m,
                                                  td[i].bn2_v, td[i].c2_b, bnS, bnT, cint);
    gemm_nn(stream, tdt, cint, wbuf, cint, nxt, cint, bnT, bnS, cBN, cint, cint, 1);
    { half_t* tmp = cur; cur = nxt; nxt = tmp; }
    dp = cint;
    run_tblock(tbs[i]);
  }

  // ---------------- transformer phase (reuse point-phase scratch) ----------
  off = mark;
  const int MT = cB * cSP;  // 4224 padded token rows
  half_t* hA    = (half_t*)alloc((size_t)MT * cDIM * 2);
  half_t* hB    = (half_t*)alloc((size_t)MT * cDIM * 2);
  half_t* lnb   = (half_t*)alloc((size_t)MT * cDIM * 2);
  half_t* attnb = (half_t*)alloc((size_t)MT * cDIM * 2);
  half_t* qkvb  = (half_t*)alloc((size_t)MT * 1536 * 2);
  float*  scor  = (float*) alloc((size_t)cB * cH * cSP * cSP * 4);
  half_t* prob  = (half_t*)alloc((size_t)cB * cH * cSP * cSP * 2);
  half_t* ffb   = (half_t*)alloc((size_t)MT * 256 * 2);
  half_t* cmp   = (half_t*)alloc((size_t)128 * cDIM * 2);
  half_t* t1    = (half_t*)alloc((size_t)128 * 256 * 2);
  half_t* t2    = (half_t*)alloc((size_t)128 * 32 * 2);
  half_t* t3    = (half_t*)alloc((size_t)128 * 64 * 2);
  float*  ypad  = (float*) alloc((size_t)128 * 16 * 4);

  k_hinit<<<cdiv(cB * cSP * cDIM, 256), 256, 0, stream>>>(jn, pos_embed, cur, hA);

  half_t* h = hA; half_t* hn = hB;
  const long long spTok = (long long)cSP * 1536;
  const long long spSq  = (long long)cSP * cSP;
  for (int L = 0; L < cDEPTH; ++L) {
    k_layernorm<<<MT, 128, 0, stream>>>(h, lnb, ly[L].ln1_g, ly[L].ln1_b, cDIM);
    cvt(ly[L].qkv_w, 512, 1536, 512, 1536);
    gemm_nn(stream, lnb, 512, wbuf, 1536, qkvb, 1536, nullptr, nullptr, MT, 1536, 512, 0);
    // scores[b,h] = SCALE * Q Kt   (batched over grid.z = B*H, B-transposed)
    launch_gemm(stream, qkvb, 1536, spTok, 128,
                qkvb + 512, 1536, spTok, 128, 1,
                scor, cSP, (long long)cH * spSq, spSq, 1,
                nullptr, 0, nullptr, nullptr,
                cSP, cSP, 128, cB * cH, cH, cISQ, 0);
    k_attn_softmax<<<cB * cH * cSP, 128, 0, stream>>>(scor, prob);
    // attn[b,h] = P V
    launch_gemm(stream, prob, cSP, (long long)cH * spSq, spSq,
                qkvb + 1024, 1536, spTok, 128, 0,
                attnb, 512, (long long)cSP * 512, 128, 0,
                nullptr, 0, nullptr, nullptr,
                cSP, 128, cSP, cB * cH, cH, 1.f, 0);
    cvt(ly[L].out_w, 512, 512, 512, 512);
    gemm_nn(stream, attnb, 512, wbuf, 512, hn, 512, ly[L].out_b, nullptr,
            MT, 512, 512, 0, h, 512);
    { half_t* tmp = h; h = hn; hn = tmp; }
    k_layernorm<<<MT, 128, 0, stream>>>(h, lnb, ly[L].ln2_g, ly[L].ln2_b, cDIM);
    cvt(ly[L].ff1_w, 512, 256, 512, 256);
    gemm_nn(stream, lnb, 512, wbuf, 256, ffb, 256, ly[L].ff1_b, nullptr, MT, 256, 512, 2);
    cvt(ly[L].ff2_w, 256, 512, 256, 512);
    gemm_nn(stream, ffb, 256, wbuf, 512, hn, 512, ly[L].ff2_b, nullptr,
            MT, 512, 256, 0, h, 512);
    { half_t* tmp = h; h = hn; hn = tmp; }
  }

  // ---------------- regression heads on first NP tokens ----------------
  k_gather68<<<cdiv(128 * cDIM, 256), 256, 0, stream>>>(h, cmp);
  cvt(fc2a_w, 512, 256, 512, 256);
  gemm_nn(stream, cmp, 512, wbuf, 256, t1, 256, fc2a_b, nullptr, 128, 256, 512, 1);
  cvt(fc2b_w, 256, 32, 256, 32);
  gemm_nn(stream, t1, 256, wbuf, 32, t2, 32, fc2b_b, nullptr, 128, 32, 256, 1); // y=relu(feat)
  cvt(fc3a_w, 32, 64, 32, 64);
  gemm_nn(stream, t2, 32, wbuf, 64, t3, 64, fc3a_b, nullptr, 128, 64, 32, 1);
  cvt(fc3b_w, 64, 3, 64, 16);
  launch_gemm(stream, t3, 64, 0, 0, wbuf, 16, 0, 0, 0,
              ypad, 16, 0, 0, 1, nullptr, 0, fc3b_b, nullptr,
              128, 3, 64, 1, 1, 1.f, 0);
  k_ycopy<<<cdiv(cB * cNP * 3, 256), 256, 0, stream>>>(ypad, (float*)d_out);
}
